// TransformerBlock_71064528880181
// MI455X (gfx1250) — matrix operand order repaired and checked by reading, not compiled, not run on hardware
//
#include <hip/hip_runtime.h>
#include <math.h>

// ---------------------------------------------------------------------------
// CDNA5 wave32 WMMA transformer block (MI455X / gfx1250).
//   v_wmma_f32_16x16x32_bf16:  D(16x16 f32) = A(16x32 bf16) * B(32x16 bf16) + C
// Fragment layouts (CDNA5 ISA 7.12.2):
//   A: lane L<16 -> row M=L, elems 0..7 = K k0+0..7, elems 8..15 = K k0+16..23
//      lane L>=16 -> row M=L-16, elems 0..7 = K k0+8..15, 8..15 = K k0+24..31
//   B: lane L<16 -> col N=L, elems 0..7 = K k0+0..7, elems 8..15 = K k0+16..23
//      lane L>=16 -> col N=L-16, elems 0..7 = K k0+8..15, 8..15 = K k0+24..31
//   C/D f32: vgpr i, lane L<16 -> (M=i, N=L); lane L>=16 -> (M=i+8, N=L-16)
// ---------------------------------------------------------------------------

typedef __bf16 bf16;
typedef __bf16 v16bf __attribute__((ext_vector_type(16)));
typedef __bf16 v8bf  __attribute__((ext_vector_type(8)));
typedef float  v8f   __attribute__((ext_vector_type(8)));

static __device__ __forceinline__ v8f wmma_bf16(v16bf a, v16bf b, v8f c) {
  return __builtin_amdgcn_wmma_f32_16x16x32_bf16(false, a, false, b, (short)0, c,
                                                 false, false);
}

// A fragment: tile origin at `base` (= &A[m0*ld + k0]), row-major, leading dim ld.
static __device__ __forceinline__ v16bf load_a_frag(const bf16* base, int ld, int lane) {
  int lh = lane & 15, hi = lane >> 4;
  const bf16* p = base + (size_t)lh * ld + hi * 8;
  union { v16bf v; v8bf h[2]; } u;
  u.h[0] = *(const v8bf*)(p);        // K = k0 + hi*8 + 0..7
  u.h[1] = *(const v8bf*)(p + 16);   // K = k0 + 16 + hi*8 + 0..7
  return u.v;
}

// B fragment: Bt stored [N][K]; tile origin at `base` (= &Bt[n0*ld + k0]).
static __device__ __forceinline__ v16bf load_b_frag(const bf16* base, int ld, int lane) {
  int lh = lane & 15, hi = lane >> 4;
  const bf16* p = base + (size_t)lh * ld + hi * 8;  // K = hi*8+0..7, 16+hi*8+0..7
  union { v16bf v; v8bf h[2]; } u;
  u.h[0] = *(const v8bf*)(p);
  u.h[1] = *(const v8bf*)(p + 16);
  return u.v;
}

// ---------------------------------------------------------------------------
// bf16 WMMA GEMM, 64x64 register tile per wave (16 f32 accumulators):
//   Out[M,N] = A[M,K] * Bt[N,K]^T (+bias)(+res)(relu)
// block = 256 threads = 8 waves arranged 2(m) x 4(n) -> 128 x 256 per block.
// Epilogue fully specialized at compile time:
//   MODE 0: bf16 out            MODE 1: f32 out
//   MODE 2: f32 + bf16 out      MODE 3: bf16 transposed [B,H,hs,S], S=2048
// ---------------------------------------------------------------------------
template <bool BIAS, bool RES, bool RELU, int MODE>
__global__ __launch_bounds__(256) void gemm_bf16_kernel(
    const bf16* __restrict__ A, const bf16* __restrict__ Bt,
    const float* __restrict__ bias, const float* __restrict__ Res,
    float* __restrict__ OutF, bf16* __restrict__ OutB, bf16* __restrict__ OutBT,
    int M, int N, int K) {
  const int wave = threadIdx.x >> 5, lane = threadIdx.x & 31;
  const int lh = lane & 15, hi = lane >> 4;
  const int wm = wave & 1, wn = wave >> 1;
  const size_t m0 = (size_t)blockIdx.y * 128 + wm * 64;
  const size_t n0 = (size_t)blockIdx.x * 256 + wn * 64;

  v8f acc[4][4];
#pragma unroll
  for (int mi = 0; mi < 4; ++mi)
#pragma unroll
    for (int nj = 0; nj < 4; ++nj) acc[mi][nj] = (v8f){0,0,0,0,0,0,0,0};

  for (int k = 0; k < K; k += 32) {
    v16bf a[4];
#pragma unroll
    for (int mi = 0; mi < 4; ++mi)
      a[mi] = load_a_frag(A + (m0 + mi * 16) * (size_t)K + k, K, lane);
#pragma unroll
    for (int nj = 0; nj < 4; ++nj) {
      v16bf b = load_b_frag(Bt + (n0 + nj * 16) * (size_t)K + k, K, lane);
#pragma unroll
      for (int mi = 0; mi < 4; ++mi) acc[mi][nj] = wmma_bf16(a[mi], b, acc[mi][nj]);
    }
  }

#pragma unroll
  for (int nj = 0; nj < 4; ++nj) {
    const size_t n = n0 + nj * 16 + lh;
    const float bv = BIAS ? bias[n] : 0.0f;
#pragma unroll
    for (int mi = 0; mi < 4; ++mi) {
      const size_t row0 = m0 + mi * 16 + hi * 8;
#pragma unroll
      for (int i = 0; i < 8; ++i) {
        const size_t row = row0 + i;
        const size_t idx = row * (size_t)N + n;
        float v = acc[mi][nj][i] + bv;
        if (RES)  v += Res[idx];
        if (RELU) v = fmaxf(v, 0.0f);
        if (MODE == 0) {
          OutB[idx] = (bf16)v;
        } else if (MODE == 1) {
          OutF[idx] = v;
        } else if (MODE == 2) {
          OutF[idx] = v;
          OutB[idx] = (bf16)v;
        } else {  // MODE 3: [B,H,hs,S] layout for attention V (S = 2048)
          const size_t bidx = row >> 11, s = row & 2047;
          OutBT[((bidx * (size_t)N + n) << 11) + s] = (bf16)v;
        }
      }
    }
  }
}

// ---------------------------------------------------------------------------
// Flash attention, causal. q,k in [B,S,H*64] bf16 (head-concat), vT in
// [B,H,64,S] bf16, out z in [B,S,H*64] bf16. One wave per 16-row query tile.
// No block barriers: waves have different causal trip counts.
// ---------------------------------------------------------------------------
__global__ __launch_bounds__(256) void attn_kernel(
    const bf16* __restrict__ q, const bf16* __restrict__ k,
    const bf16* __restrict__ vt, bf16* __restrict__ z, int S, int H) {
  const int wave = threadIdx.x >> 5, lane = threadIdx.x & 31;
  const int lh = lane & 15, hi = lane >> 4;
  const int bh = blockIdx.y, b = bh / H, h = bh % H;
  const int D = H * 64;
  const int m0 = blockIdx.x * 128 + wave * 16;

  const bf16* qb = q + (size_t)b * S * D + h * 64;
  const bf16* kb = k + (size_t)b * S * D + h * 64;
  const bf16* vb = vt + (size_t)bh * 64 * S;

  __shared__ __align__(32) bf16 Plds[8][16 * 32];
  bf16* Pw = Plds[wave];

  // Q fragments for the two hs chunks (hs = 64 = 2 * K32)
  const v16bf aq0 = load_a_frag(qb + (size_t)m0 * D + 0, D, lane);
  const v16bf aq1 = load_a_frag(qb + (size_t)m0 * D + 32, D, lane);

  v8f o[4];
#pragma unroll
  for (int j = 0; j < 4; ++j) o[j] = (v8f){0,0,0,0,0,0,0,0};
  float mrow[8], lrow[8];
#pragma unroll
  for (int i = 0; i < 8; ++i) { mrow[i] = -__builtin_inff(); lrow[i] = 0.0f; }

  const float scale = 0.125f;  // 1/sqrt(64)
  const int rowbase = m0 + hi * 8;

  for (int kc = 0; kc < m0 + 16; kc += 32) {
    // ---- scores: two 16x16 tiles over keys [kc,kc+16) and [kc+16,kc+32)
    v8f s0 = (v8f){0,0,0,0,0,0,0,0}, s1 = s0;
    s0 = wmma_bf16(aq0, load_b_frag(kb + (size_t)kc * D + 0,  D, lane), s0);
    s0 = wmma_bf16(aq1, load_b_frag(kb + (size_t)kc * D + 32, D, lane), s0);
    s1 = wmma_bf16(aq0, load_b_frag(kb + (size_t)(kc + 16) * D + 0,  D, lane), s1);
    s1 = wmma_bf16(aq1, load_b_frag(kb + (size_t)(kc + 16) * D + 32, D, lane), s1);

    // ---- scale + causal mask + online softmax
    float p0[8], p1[8], cm[8], rs[8], scl[8];
#pragma unroll
    for (int i = 0; i < 8; ++i) {
      const int row = rowbase + i;
      float x0 = s0[i] * scale, x1 = s1[i] * scale;
      if (kc + lh      > row) x0 = -__builtin_inff();
      if (kc + 16 + lh > row) x1 = -__builtin_inff();
      p0[i] = x0; p1[i] = x1;
      cm[i] = fmaxf(x0, x1);
    }
#pragma unroll
    for (int off = 1; off < 16; off <<= 1)
#pragma unroll
      for (int i = 0; i < 8; ++i) cm[i] = fmaxf(cm[i], __shfl_xor(cm[i], off, 32));
#pragma unroll
    for (int i = 0; i < 8; ++i) {
      const float mn = fmaxf(mrow[i], cm[i]);
      scl[i] = __expf(mrow[i] - mn);   // 0 on first chunk (exp(-inf))
      mrow[i] = mn;
      p0[i] = __expf(p0[i] - mn);
      p1[i] = __expf(p1[i] - mn);
      rs[i] = p0[i] + p1[i];
    }
#pragma unroll
    for (int off = 1; off < 16; off <<= 1)
#pragma unroll
      for (int i = 0; i < 8; ++i) rs[i] += __shfl_xor(rs[i], off, 32);
#pragma unroll
    for (int i = 0; i < 8; ++i) {
      lrow[i] = lrow[i] * scl[i] + rs[i];
#pragma unroll
      for (int j = 0; j < 4; ++j) o[j][i] *= scl[i];
      // stage P as bf16, row-major [16][32], for re-read as an A fragment
      Pw[(i + hi * 8) * 32 + lh]      = (bf16)p0[i];
      Pw[(i + hi * 8) * 32 + 16 + lh] = (bf16)p1[i];
    }
    // LDS ops from one wave complete in order (DScnt in-order); stop compiler
    // reordering and drain the DS counter before the cross-lane re-read.
    asm volatile("s_wait_dscnt 0" ::: "memory");

    // ---- P(16x32) * V(32x64)
    v16bf ap = load_a_frag(Pw, 32, lane);
#pragma unroll
    for (int j = 0; j < 4; ++j)
      o[j] = wmma_bf16(ap, load_b_frag(vb + (size_t)(j * 16) * S + kc, S, lane), o[j]);
  }

  // ---- normalize and emit bf16 (head-concat layout)
  float inv[8];
#pragma unroll
  for (int i = 0; i < 8; ++i) inv[i] = 1.0f / lrow[i];
#pragma unroll
  for (int j = 0; j < 4; ++j)
#pragma unroll
    for (int i = 0; i < 8; ++i) {
      const size_t row = (size_t)b * S + m0 + i + hi * 8;
      z[row * D + h * 64 + j * 16 + lh] = (bf16)(o[j][i] * inv[i]);
    }
}

// ---------------------------------------------------------------------------
// Data-prep kernels
// ---------------------------------------------------------------------------
__global__ void cvt_f32_bf16(const float* __restrict__ in, bf16* __restrict__ out,
                             size_t n) {
  size_t i = (size_t)blockIdx.x * blockDim.x + threadIdx.x;
  if (i < n) out[i] = (bf16)in[i];
}

// in[z][r][c] (f32) -> out[z][c][r] (bf16)
__global__ void transpose_cvt(const float* __restrict__ in, bf16* __restrict__ out,
                              int R, int C) {
  size_t base = (size_t)blockIdx.z * R * C;
  size_t i = (size_t)blockIdx.x * blockDim.x + threadIdx.x;
  if (i >= (size_t)R * C) return;
  int r = (int)(i / C), c = (int)(i % C);
  out[base + (size_t)c * R + r] = (bf16)in[base + i];
}

// ---------------------------------------------------------------------------
extern "C" void kernel_launch(void* const* d_in, const int* in_sizes, int n_in,
                              void* d_out, int out_size, void* d_ws, size_t ws_size,
                              hipStream_t stream) {
  const int B = 4, S = 2048, D = 768, H = 12, F = 3072;
  const size_t MS = (size_t)B * S;  // 8192 rows

  const float* x  = (const float*)d_in[0];
  const float* Wq = (const float*)d_in[1];
  const float* Wk = (const float*)d_in[2];
  const float* Wv = (const float*)d_in[3];
  const float* Wo = (const float*)d_in[4];
  const float* bo = (const float*)d_in[5];
  const float* W1 = (const float*)d_in[6];
  const float* b1 = (const float*)d_in[7];
  const float* W2 = (const float*)d_in[8];
  const float* b2 = (const float*)d_in[9];

  // ---- workspace carve (256B aligned) ----
  char* w = (char*)d_ws;
  auto alloc = [&](size_t bytes) -> void* {
    void* p = (void*)w;
    w += (bytes + 255) & ~(size_t)255;
    return p;
  };
  bf16* wqt = (bf16*)alloc((size_t)D * D * 2);
  bf16* wkt = (bf16*)alloc((size_t)D * D * 2);
  bf16* wvt = (bf16*)alloc((size_t)D * D * 2);
  bf16* wot = (bf16*)alloc((size_t)D * D * 2);
  bf16* w1t = (bf16*)alloc((size_t)D * F * 2);
  bf16* w2t = (bf16*)alloc((size_t)F * D * 2);
  bf16* xb  = (bf16*)alloc(MS * D * 2);
  bf16* qb  = (bf16*)alloc(MS * D * 2);   // | contiguous 4*MS*D*2 = MS*F*2:
  bf16* kbuf= (bf16*)alloc(MS * D * 2);   // | reused as FFN hidden after attn
  bf16* vtb = (bf16*)alloc(MS * D * 2);   // | (zb dead after O-proj, which
  bf16* zb  = (bf16*)alloc(MS * D * 2);   // |  precedes FFN1)
  float* yf = (float*)alloc(MS * D * 4);
  bf16* yb  = xb;  // x_bf16 dead after QKV GEMMs
  bf16* hb  = qb;  // q,k,vT,z region reused as FFN hidden [8192,3072]

  const dim3 blk(256);

  // ---- precision conversion + weight transposes ----
  cvt_f32_bf16<<<dim3((unsigned)((MS * D + 255) / 256)), blk, 0, stream>>>(x, xb, MS * D);
  transpose_cvt<<<dim3((D * 64 + 255) / 256, 1, H), blk, 0, stream>>>(Wq, wqt, D, 64);
  transpose_cvt<<<dim3((D * 64 + 255) / 256, 1, H), blk, 0, stream>>>(Wk, wkt, D, 64);
  transpose_cvt<<<dim3((D * 64 + 255) / 256, 1, H), blk, 0, stream>>>(Wv, wvt, D, 64);
  transpose_cvt<<<dim3((D * D + 255) / 256, 1, 1), blk, 0, stream>>>(Wo, wot, D, D);
  transpose_cvt<<<dim3((D * F + 255) / 256, 1, 1), blk, 0, stream>>>(W1, w1t, D, F);
  transpose_cvt<<<dim3((F * D + 255) / 256, 1, 1), blk, 0, stream>>>(W2, w2t, F, D);

  // ---- QKV projections: [8192,768] x [768,768] (per-head weights concat on N)
  const dim3 gD(D / 256, (unsigned)(MS / 128));
  gemm_bf16_kernel<false, false, false, 0><<<gD, blk, 0, stream>>>(
      xb, wqt, nullptr, nullptr, nullptr, qb, nullptr, (int)MS, D, D);
  gemm_bf16_kernel<false, false, false, 0><<<gD, blk, 0, stream>>>(
      xb, wkt, nullptr, nullptr, nullptr, kbuf, nullptr, (int)MS, D, D);
  // V projection writes transposed [B,H,hs,S] directly from the epilogue
  gemm_bf16_kernel<false, false, false, 3><<<gD, blk, 0, stream>>>(
      xb, wvt, nullptr, nullptr, nullptr, nullptr, vtb, (int)MS, D, D);

  // ---- causal flash attention ----
  attn_kernel<<<dim3(S / 128, B * H), blk, 0, stream>>>(qb, kbuf, vtb, zb, S, H);

  // ---- output projection + bias + residual(x): y (f32 + bf16) ----
  gemm_bf16_kernel<true, true, false, 2><<<gD, blk, 0, stream>>>(
      zb, wot, bo, x, yf, yb, nullptr, (int)MS, D, D);

  // ---- FFN1 + ReLU -> h (bf16) ----
  gemm_bf16_kernel<true, false, true, 0><<<dim3(F / 256, (unsigned)(MS / 128)), blk, 0, stream>>>(
      yb, w1t, b1, nullptr, nullptr, hb, nullptr, (int)MS, F, D);

  // ---- FFN2 + bias + residual(y) -> d_out (f32) ----
  gemm_bf16_kernel<true, true, false, 1><<<gD, blk, 0, stream>>>(
      hb, w2t, b2, yf, (float*)d_out, nullptr, nullptr, (int)MS, D, F);
}
